// TokenFormerBlock_66228395704329
// MI455X (gfx1250) — compile-verified, matrix operand order checked by reading, not run on hardware
//
#include <hip/hip_runtime.h>

typedef __attribute__((ext_vector_type(16))) _Float16 v16h;
typedef __attribute__((ext_vector_type(8)))  _Float16 v8h;
typedef __attribute__((ext_vector_type(4)))  _Float16 v4h;
typedef __attribute__((ext_vector_type(8)))  float    v8f;
typedef __attribute__((ext_vector_type(4)))  float    v4f;
typedef int v4i __attribute__((vector_size(16)));

#define EMB   1024
#define HEADS 16
#define HD    64
#define NEG_INF (-__builtin_inff())

// ---------------------------------------------------------------------------
// gfx1250 async global->LDS copy (ASYNCcnt).
// ---------------------------------------------------------------------------
#if defined(__has_builtin)
#  if __has_builtin(__builtin_amdgcn_global_load_async_to_lds_b128) && \
      __has_builtin(__builtin_amdgcn_s_wait_asynccnt)
#    define HAS_ASYNC_LDS 1
#  endif
#endif
#ifndef HAS_ASYNC_LDS
#  define HAS_ASYNC_LDS 0
#endif

__device__ __forceinline__ void copy16_to_lds(void* lds, const void* g) {
#if HAS_ASYNC_LDS
    __builtin_amdgcn_global_load_async_to_lds_b128(
        (__attribute__((address_space(1))) v4i*)(g),
        (__attribute__((address_space(3))) v4i*)(lds), 0, 0);
#else
    *(v8h*)lds = *(const v8h*)g;
#endif
}
__device__ __forceinline__ void lds_copy_wait() {
#if HAS_ASYNC_LDS
    __builtin_amdgcn_s_wait_asynccnt(0);
#endif
}

// Gather a 16-element f16 WMMA fragment from one LDS row.
// ISA 16-bit A/B layout: lanes 0-15 -> K = {0..7, 16..23}; lanes 16-31 -> K = {8..15, 24..31}.
__device__ __forceinline__ v16h load_frag(const _Float16* row, int khalf) {
    v8h lo = *reinterpret_cast<const v8h*>(row + khalf);
    v8h hi = *reinterpret_cast<const v8h*>(row + khalf + 16);
    v16h r;
#pragma unroll
    for (int i = 0; i < 8; ++i) { r[i] = lo[i]; r[i + 8] = hi[i]; }
    return r;
}

// ---------------------------------------------------------------------------
// f32 -> f16 conversion (weights). One thread = 4 elements, launch n/1024 WGs.
// ---------------------------------------------------------------------------
__global__ __launch_bounds__(256) void cvt_f16_kernel(const float* __restrict__ X,
                                                      _Float16* __restrict__ Y) {
    const size_t i = ((size_t)blockIdx.x * 256 + threadIdx.x) * 4;
    v4f x = *(const v4f*)(X + i);
    v4h y;
#pragma unroll
    for (int j = 0; j < 4; ++j) y[j] = (_Float16)x[j];
    *(v4h*)(Y + i) = y;
}

// ---------------------------------------------------------------------------
// LayerNorm over EMB=1024 (no affine), f32 in -> f16 out. 256 thr / row.
// ---------------------------------------------------------------------------
__global__ __launch_bounds__(256) void layernorm_kernel(const float* __restrict__ X,
                                                        _Float16* __restrict__ Y) {
    __shared__ float red[256];
    const int row = blockIdx.x, tid = threadIdx.x;
    const v4f xv = *(const v4f*)(X + (size_t)row * EMB + tid * 4);
    float s = xv[0] + xv[1] + xv[2] + xv[3];
    red[tid] = s; __syncthreads();
    for (int st = 128; st > 0; st >>= 1) { if (tid < st) red[tid] += red[tid + st]; __syncthreads(); }
    const float mu = red[0] * (1.0f / EMB);
    __syncthreads();
    float vs = 0.f;
#pragma unroll
    for (int i = 0; i < 4; ++i) { float d = xv[i] - mu; vs += d * d; }
    red[tid] = vs; __syncthreads();
    for (int st = 128; st > 0; st >>= 1) { if (tid < st) red[tid] += red[tid + st]; __syncthreads(); }
    const float inv = rsqrtf(red[0] * (1.0f / EMB) + 1e-5f);
    v4h y;
#pragma unroll
    for (int i = 0; i < 4; ++i) y[i] = (_Float16)((xv[i] - mu) * inv);
    *(v4h*)(Y + (size_t)row * EMB + tid * 4) = y;
}

// ---------------------------------------------------------------------------
// Row softmax in place on f16 scores (stats in f32). 256 thr / row.
// ---------------------------------------------------------------------------
__global__ __launch_bounds__(256) void softmax16_kernel(_Float16* __restrict__ S, int N) {
    __shared__ float red[256];
    const int row = blockIdx.x, tid = threadIdx.x;
    const int E = N >> 8;                       // 4 or 16 contiguous halves per thread
    _Float16* p = S + (size_t)row * N + tid * E;
    float m = NEG_INF;
    for (int j = 0; j < E; ++j) m = fmaxf(m, (float)p[j]);
    red[tid] = m; __syncthreads();
    for (int st = 128; st > 0; st >>= 1) { if (tid < st) red[tid] = fmaxf(red[tid], red[tid + st]); __syncthreads(); }
    m = red[0]; __syncthreads();
    float sum = 0.f;
    for (int j = 0; j < E; ++j) { float e = __expf((float)p[j] - m); p[j] = (_Float16)e; sum += e; }
    red[tid] = sum; __syncthreads();
    for (int st = 128; st > 0; st >>= 1) { if (tid < st) red[tid] += red[tid + st]; __syncthreads(); }
    const float inv = 1.0f / red[0];
    for (int j = 0; j < E; ++j) p[j] = (_Float16)((float)p[j] * inv);
}

// ---------------------------------------------------------------------------
// WMMA GEMM, f16 operands, f32 accumulate. 256x128 WG tile, BK=64,
// double-buffered LDS + async staging pipeline; MODE-1 B tile goes through a
// register relay so compute never waits on staging loads.
// 8 waves (4x2), 64x64 per wave -> 32 v_wmma per wave per K-stage.
// MODE 0: D[m][n] = scale * sum_k A[m*K+k] * Bm[n*K+k]   -> f16 scores
// MODE 1: D[m][n] = sum_k A[m*K+k] * Bm[k*N+n] (+res)    -> f16 (OUT16) or f32
// ---------------------------------------------------------------------------
template<int MODE, bool OUT16>
__global__ __launch_bounds__(256) void gemm_kernel(const _Float16* __restrict__ A,
                                                   const _Float16* __restrict__ Bm,
                                                   const float* __restrict__ res,
                                                   void* __restrict__ Dout,
                                                   int M, int N, int K, float scale) {
    constexpr int BM = 256, BN = 128, BK = 64;
    constexpr int LDR = BK + 8;   // 72 halves: 16B-aligned chunks, conflict padding
    __shared__ alignas(16) _Float16 lsA[2][BM][LDR];
    __shared__ alignas(16) _Float16 lsB[2][BN][LDR];

    const int tid   = threadIdx.x;
    const int lane  = tid & 31;
    const int wave  = tid >> 5;
    const int wm    = wave >> 1;       // 0..3
    const int wn    = wave & 1;        // 0..1
    const int m0    = blockIdx.x * BM;
    const int n0    = blockIdx.y * BN;
    const int l15   = lane & 15;
    const int khalf = (lane >> 4) * 8;
    const int hi8   = khalf;

    v8f acc[4][4];
#pragma unroll
    for (int r = 0; r < 4; ++r)
#pragma unroll
        for (int c = 0; c < 4; ++c) acc[r][c] = {};

    // Async staging of A (always) and B (MODE 0 only)
    auto stage_async = [&](int buf, int k0) {
#pragma unroll
        for (int i = 0; i < 8; ++i) {
            int hidx = (tid + i * 256) * 8;
            int r = hidx >> 6, c = hidx & 63;
            copy16_to_lds(&lsA[buf][r][c], A + (size_t)(m0 + r) * K + k0 + c);
        }
        if (MODE == 0) {
#pragma unroll
            for (int i = 0; i < 4; ++i) {
                int hidx = (tid + i * 256) * 8;
                int r = hidx >> 6, c = hidx & 63;
                copy16_to_lds(&lsB[buf][r][c], Bm + (size_t)(n0 + r) * K + k0 + c);
            }
        }
    };
    // MODE 1 register relay: load B rows now, scatter transposed later
    auto loadB_regs = [&](int k0, v8h* breg) {
#pragma unroll
        for (int i = 0; i < 4; ++i) {
            int hidx = (tid + i * 256) * 8;
            int kr = hidx >> 7, cc = hidx & 127;
            breg[i] = *(const v8h*)(Bm + (size_t)(k0 + kr) * N + n0 + cc);
        }
    };
    auto storeB_lds = [&](int buf, const v8h* breg) {
#pragma unroll
        for (int i = 0; i < 4; ++i) {
            int hidx = (tid + i * 256) * 8;
            int kr = hidx >> 7, cc = hidx & 127;
#pragma unroll
            for (int j = 0; j < 8; ++j) lsB[buf][cc + j][kr] = breg[i][j];
        }
    };

    auto compute = [&](int buf) {
#pragma unroll
        for (int ks = 0; ks < 2; ++ks) {
            v16h af[4], bf[4];
#pragma unroll
            for (int r = 0; r < 4; ++r)
                af[r] = load_frag(&lsA[buf][wm * 64 + r * 16 + l15][ks * 32], khalf);
#pragma unroll
            for (int c = 0; c < 4; ++c)
                bf[c] = load_frag(&lsB[buf][wn * 64 + c * 16 + l15][ks * 32], khalf);
#pragma unroll
            for (int r = 0; r < 4; ++r)
#pragma unroll
                for (int c = 0; c < 4; ++c)
                    acc[r][c] = __builtin_amdgcn_wmma_f32_16x16x32_f16(
                        false, af[r], false, bf[c], (short)0, acc[r][c], false, false);
        }
    };

    v8h breg[4];
    stage_async(0, 0);
    if (MODE == 1) { loadB_regs(0, breg); storeB_lds(0, breg); }
    lds_copy_wait();
    __syncthreads();
    int buf = 0;
    for (int k0 = 0; k0 < K; k0 += BK) {
        const bool more = (k0 + BK) < K;
        if (more) {
            stage_async(buf ^ 1, k0 + BK);          // async copies overlap compute
            if (MODE == 1) loadB_regs(k0 + BK, breg); // global latency overlaps compute
        }
        compute(buf);
        if (more && MODE == 1) storeB_lds(buf ^ 1, breg);
        lds_copy_wait();
        __syncthreads();
        buf ^= 1;
    }

    // Epilogue: D-tile layout -> row = v + 8*(lane>=16), col = lane&15
#pragma unroll
    for (int r = 0; r < 4; ++r)
#pragma unroll
        for (int c = 0; c < 4; ++c)
#pragma unroll
            for (int v = 0; v < 8; ++v) {
                int row = m0 + wm * 64 + r * 16 + v + hi8;
                int col = n0 + wn * 64 + c * 16 + l15;
                float val = acc[r][c][v];
                if (MODE == 0) {
                    ((_Float16*)Dout)[(size_t)row * N + col] = (_Float16)(val * scale);
                } else if (OUT16) {
                    ((_Float16*)Dout)[(size_t)row * N + col] = (_Float16)val;
                } else {
                    if (res) val += res[(size_t)row * N + col];
                    ((float*)Dout)[(size_t)row * N + col] = val;
                }
            }
}

// ---------------------------------------------------------------------------
// Flash-style causal SDPA, f16 in/out. Grid: (T/64, B*HEADS), 256 threads.
// 64 query rows / WG; 32-key blocks double-buffered: K via async-to-LDS,
// V via register relay into a transposed LDS tile so the PV B-fragment is
// two aligned ds_load_b128 (no strided gathers). Softmax 4 threads/row.
// ---------------------------------------------------------------------------
__global__ __launch_bounds__(256) void sdpa_kernel(const _Float16* __restrict__ Q,
                                                   const _Float16* __restrict__ Kg,
                                                   const _Float16* __restrict__ Vg,
                                                   _Float16* __restrict__ O,
                                                   int T) {
    __shared__ alignas(16) _Float16 Qs[64][72];
    __shared__ alignas(16) _Float16 Ks[2][32][72];
    __shared__ alignas(16) _Float16 VsT[2][64][40];   // [dcol][key] transposed
    __shared__ float    Ss[64][36];
    __shared__ alignas(16) _Float16 Ps[64][40];
    __shared__ float    part4[64][4];
    __shared__ float    mrow[64], lrow[64], rrow[64];

    const int tid = threadIdx.x, lane = tid & 31, wave = tid >> 5;
    const int qb = blockIdx.x;
    const int b  = blockIdx.y >> 4, h = blockIdx.y & 15;
    const int q0 = qb * 64;
    const size_t baseRow = (size_t)b * T;
    const int colOff = h * HD;
    const int l15   = lane & 15;
    const int khalf = (lane >> 4) * 8;
    const int hi8   = khalf;

    const int vkr = (tid * 8) >> 6;        // this thread's V row (key) 0..31
    const int vcc = (tid * 8) & 63;        // this thread's V col base

    auto stageK = [&](int buf, int kb) {
        const int k0 = kb * 32;
        copy16_to_lds(&Ks[buf][vkr][vcc], Kg + (baseRow + k0 + vkr) * EMB + colOff + vcc);
    };
    auto loadV_reg = [&](int kb) {
        const int k0 = kb * 32;
        return *(const v8h*)(Vg + (baseRow + k0 + vkr) * EMB + colOff + vcc);
    };
    auto storeV_lds = [&](int buf, v8h vv) {
#pragma unroll
        for (int j = 0; j < 8; ++j) VsT[buf][vcc + j][vkr] = vv[j];
    };

    // Stage Q tile (64 x 64 halves) + first K/V block
#pragma unroll
    for (int i = 0; i < 2; ++i) {
        int hidx = (tid + i * 256) * 8;
        int r = hidx >> 6, d = hidx & 63;
        copy16_to_lds(&Qs[r][d], Q + (baseRow + q0 + r) * EMB + colOff + d);
    }
    stageK(0, 0);
    storeV_lds(0, loadV_reg(0));
    if (tid < 64) { mrow[tid] = NEG_INF; lrow[tid] = 0.f; }
    v8f o0 = {}, o1 = {};
    lds_copy_wait();
    __syncthreads();

    const int kbmax = 2 * qb + 1;
    int buf = 0;
    for (int kb = 0; kb <= kbmax; ++kb) {
        const int k0 = kb * 32;
        const bool more = kb < kbmax;
        v8h vnext = {};
        if (more) {
            stageK(buf ^ 1, kb + 1);       // async, overlaps everything below
            vnext = loadV_reg(kb + 1);     // global latency overlaps compute
        }

        // ---- S = (Q K^T) * 0.125, causal mask, spill to LDS ----
        {
            const int rt = wave >> 1, ct = wave & 1;
            v8f s = {};
#pragma unroll
            for (int ks = 0; ks < 2; ++ks) {
                v16h aq = load_frag(&Qs[rt * 16 + l15][ks * 32], khalf);
                v16h bk = load_frag(&Ks[buf][ct * 16 + l15][ks * 32], khalf);
                s = __builtin_amdgcn_wmma_f32_16x16x32_f16(
                    false, aq, false, bk, (short)0, s, false, false);
            }
#pragma unroll
            for (int v = 0; v < 8; ++v) {
                int rl = rt * 16 + v + hi8, cl = ct * 16 + l15;
                Ss[rl][cl] = ((k0 + cl) <= (q0 + rl)) ? s[v] * 0.125f : NEG_INF;
            }
        }
        __syncthreads();

        // ---- online softmax, 4 threads per row ----
        const int srow = tid >> 2, spart = tid & 3;
        {
            float mloc = NEG_INF;
#pragma unroll
            for (int j = 0; j < 8; ++j) mloc = fmaxf(mloc, Ss[srow][spart * 8 + j]);
            part4[srow][spart] = mloc;
        }
        __syncthreads();
        if (tid < 64) {
            float mloc = fmaxf(fmaxf(part4[tid][0], part4[tid][1]),
                               fmaxf(part4[tid][2], part4[tid][3]));
            float mnew = fmaxf(mrow[tid], mloc);
            rrow[tid] = __expf(mrow[tid] - mnew);
            mrow[tid] = mnew;
        }
        __syncthreads();
        {
            const float mnew = mrow[srow];
            float sum = 0.f;
#pragma unroll
            for (int j = 0; j < 8; ++j) {
                float e = __expf(Ss[srow][spart * 8 + j] - mnew);
                Ps[srow][spart * 8 + j] = (_Float16)e;
                sum += e;
            }
            part4[srow][spart] = sum;
        }
        __syncthreads();
        if (tid < 64)
            lrow[tid] = lrow[tid] * rrow[tid] +
                        (part4[tid][0] + part4[tid][1] + part4[tid][2] + part4[tid][3]);

        // ---- O = O*resc + P @ V : wave (rh, dsl) owns rows rh*32.., cols dsl*16.. ----
        {
            const int dsl = wave & 3, rh = wave >> 2;
            v16h bfrag = load_frag(&VsT[buf][dsl * 16 + l15][0], khalf);
            v16h a0 = load_frag(&Ps[rh * 32 + l15][0], khalf);
            v16h a1 = load_frag(&Ps[rh * 32 + 16 + l15][0], khalf);
#pragma unroll
            for (int v = 0; v < 8; ++v) {
                o0[v] *= rrow[rh * 32 + v + hi8];
                o1[v] *= rrow[rh * 32 + 16 + v + hi8];
            }
            o0 = __builtin_amdgcn_wmma_f32_16x16x32_f16(false, a0, false, bfrag, (short)0, o0, false, false);
            o1 = __builtin_amdgcn_wmma_f32_16x16x32_f16(false, a1, false, bfrag, (short)0, o1, false, false);
        }
        if (more) storeV_lds(buf ^ 1, vnext);
        lds_copy_wait();
        __syncthreads();
        buf ^= 1;
    }

    {
        const int dsl = wave & 3, rh = wave >> 2;
#pragma unroll
        for (int v = 0; v < 8; ++v) {
            int r0 = rh * 32 + v + hi8, r1 = r0 + 16;
            int col = colOff + dsl * 16 + l15;
            O[(baseRow + q0 + r0) * EMB + col] = (_Float16)(o0[v] / lrow[r0]);
            O[(baseRow + q0 + r1) * EMB + col] = (_Float16)(o1[v] / lrow[r1]);
        }
    }
}

// ---------------------------------------------------------------------------
// Host-side orchestration
// ---------------------------------------------------------------------------
extern "C" void kernel_launch(void* const* d_in, const int* in_sizes, int n_in,
                              void* d_out, int out_size, void* d_ws, size_t ws_size,
                              hipStream_t stream) {
    const float* x = (const float*)d_in[0];
    const float* W32[10];
    for (int i = 0; i < 10; ++i) W32[i] = (const float*)d_in[1 + i];
    float* out = (float*)d_out;

    const int M = 4096, C = EMB, N1 = 1024, N4 = 4096, T = 2048;
    const size_t W1 = (size_t)1024 * 1024;     // elements per small weight bank
    const size_t CH = (size_t)M * C;           // 4M elements

    // f16 workspace carving (~120 MiB total)
    _Float16* wh   = (_Float16*)d_ws;          // 16M halves: 8 small banks + fK + fV
    _Float16* Wh[10];
    for (int i = 0; i < 8; ++i) Wh[i] = wh + i * W1;
    Wh[8] = wh + 8 * W1;                       // fK: 4M halves
    Wh[9] = wh + 12 * W1;                      // fV: 4M halves
    _Float16* hh  = wh + 16 * W1;              // LN output (M x C)
    _Float16* qh  = hh + CH;
    _Float16* kh  = qh + CH;
    _Float16* vh  = kh + CH;
    _Float16* oh  = vh + CH;
    _Float16* S16 = oh + CH;                   // scores/probs, up to M x 4096
    float*    x2  = (float*)(S16 + (size_t)M * 4096);

    const float scale = 0.03125f;              // EMB^-0.5

    // Convert the 10 weight banks to f16 (read once, L2-resident afterwards)
    for (int i = 0; i < 10; ++i) {
        int n = (i < 8) ? (int)(W1) : (int)(4 * W1);
        cvt_f16_kernel<<<n / 1024, 256, 0, stream>>>(W32[i], Wh[i]);
    }

    auto patt16 = [&](const _Float16* Kw, const _Float16* Vw, int N,
                      const _Float16* src, _Float16* dst) {
        gemm_kernel<0, true><<<dim3(M / 256, N / 128), 256, 0, stream>>>(src, Kw, nullptr, S16, M, N, C, scale);
        softmax16_kernel<<<M, 256, 0, stream>>>(S16, N);
        gemm_kernel<1, true><<<dim3(M / 256, C / 128), 256, 0, stream>>>(S16, Vw, nullptr, dst, M, C, N, 1.0f);
    };
    auto patt32 = [&](const _Float16* Kw, const _Float16* Vw, int N,
                      const _Float16* src, const float* res, float* dst) {
        gemm_kernel<0, true><<<dim3(M / 256, N / 128), 256, 0, stream>>>(src, Kw, nullptr, S16, M, N, C, scale);
        softmax16_kernel<<<M, 256, 0, stream>>>(S16, N);
        gemm_kernel<1, false><<<dim3(M / 256, C / 128), 256, 0, stream>>>(S16, Vw, res, dst, M, C, N, 1.0f);
    };

    layernorm_kernel<<<M, 256, 0, stream>>>(x, hh);            // hh = LN(x)
    patt16(Wh[0], Wh[1], N1, hh, qh);                          // q
    patt16(Wh[2], Wh[3], N1, hh, kh);                          // k
    patt16(Wh[4], Wh[5], N1, hh, vh);                          // v
    sdpa_kernel<<<dim3(T / 64, 2 * HEADS), 256, 0, stream>>>(qh, kh, vh, oh, T);
    patt32(Wh[6], Wh[7], N1, oh, x, x2);                       // x2 = proj + x
    layernorm_kernel<<<M, 256, 0, stream>>>(x2, hh);           // hh = LN(x2)
    patt32(Wh[8], Wh[9], N4, hh, x2, out);                     // out = ffn + x2
}
